// QLSTM_65481071408935
// MI455X (gfx1250) — matrix operand order repaired and checked by reading, not compiled, not run on hardware
//
#include <hip/hip_runtime.h>
#include <hip/hip_bf16.h>
#include <math.h>

// ---------------------------------------------------------------------------
// QLSTM forward for MI455X (gfx1250): bf16 WMMA GEMMs + fp32 cell update.
// SEQ=512, B=128, D_IN=1024, D_H=1024, NQ=1, cin = 2048.
//
// Per-step GEMM: each wave owns one (gate, 16-col n-tile) and sweeps all 8
// batch M-tiles, so every weight byte is fetched from L2 exactly once per
// step (12 MB instead of 96 MB). The shared A (comb) k-chunk is staged in
// LDS per block and re-read with ds_load_b128; 8 independent WMMAs issue
// back-to-back per k-chunk (no D->A/B hazard).
// ---------------------------------------------------------------------------

#define SEQ   512
#define BATCH 128
#define DIN   1024
#define DH    1024
#define CIN   2048

// LDS A-stage: 128 rows x 32 bf16, row stride 40 bf16 (80 B: 16B-aligned,
// bank-conflict-free). Pre-activation exchange: 128 x 16 f32, row stride 17.
#define A_STRIDE 40
#define P_STRIDE 17
#define P_GATE   (BATCH * P_STRIDE)      // 2176 floats per gate

typedef __bf16 bf16_t;
typedef __attribute__((ext_vector_type(16))) __bf16 v16bf;
typedef __attribute__((ext_vector_type(8)))  __bf16 v8bf;
typedef __attribute__((ext_vector_type(8)))  float  v8f;

__device__ __forceinline__ float sigmoidf_(float x) {
    return 1.0f / (1.0f + __expf(-x));
}

// ---------------------------------------------------------------------------
// One-time: convert gate weights Wi/Wu/Wo (1024 x 2048 f32, row-major over k)
// into a gate-major bf16 buffer Wb[3][1024][2048].
// ---------------------------------------------------------------------------
__global__ void qlstm_cvt_w(const float* __restrict__ Wi,
                            const float* __restrict__ Wu,
                            const float* __restrict__ Wo,
                            bf16_t* __restrict__ Wb) {
    const int N = DH * CIN;
    for (int idx = blockIdx.x * blockDim.x + threadIdx.x; idx < N;
         idx += gridDim.x * blockDim.x) {
        Wb[idx]         = (bf16_t)Wi[idx];
        Wb[N + idx]     = (bf16_t)Wu[idx];
        Wb[2 * N + idx] = (bf16_t)Wo[idx];
    }
}

// ---------------------------------------------------------------------------
// One-time: zero c, h (f32) and the h-half of comb0 (bf16).
// ---------------------------------------------------------------------------
__global__ void qlstm_init(float* __restrict__ c, float* __restrict__ h,
                           bf16_t* __restrict__ comb0) {
    int i = blockIdx.x * blockDim.x + threadIdx.x;   // 0 .. 131071
    if (i < BATCH * DH) {
        c[i] = 0.0f;
        h[i] = 0.0f;
        int b = i >> 10, k = i & 1023;
        comb0[(size_t)b * CIN + DIN + k] = (bf16_t)0.0f;
    }
}

// ---------------------------------------------------------------------------
// Per-step: convert x_t (128 x 1024 f32) into comb[:, 0:1024] bf16.
// ---------------------------------------------------------------------------
__global__ void qlstm_cvt_x(const float* __restrict__ xt,
                            bf16_t* __restrict__ comb) {
    int i = blockIdx.x * blockDim.x + threadIdx.x;   // 0 .. 131071
    if (i < BATCH * DIN) {
        int b = i >> 10, k = i & 1023;
        comb[(size_t)b * CIN + k] = (bf16_t)xt[i];
    }
}

// ---------------------------------------------------------------------------
// Per-step: quantum forget gate. One block per batch row:
// f_in = dot(x_t, Wq[:1024]) + dot(h_prev, Wq[1024:]) + bq
// f    = sigmoid(cos(f_in * qlin_w) * cos(qparams))
// ---------------------------------------------------------------------------
__global__ __launch_bounds__(256)
void qlstm_qgate(const float* __restrict__ xt, const float* __restrict__ h,
                 const float* __restrict__ Wq, const float* __restrict__ bq,
                 const float* __restrict__ qlw, const float* __restrict__ qp,
                 float* __restrict__ fgate) {
    __shared__ float red[256];
    const int b = blockIdx.x;
    float s = 0.0f;
    for (int k = threadIdx.x; k < DIN; k += 256)
        s += xt[(size_t)b * DIN + k] * Wq[k];
    for (int k = threadIdx.x; k < DH; k += 256)
        s += h[(size_t)b * DH + k] * Wq[DIN + k];
    red[threadIdx.x] = s;
    __syncthreads();
    for (int off = 128; off > 0; off >>= 1) {
        if ((int)threadIdx.x < off) red[threadIdx.x] += red[threadIdx.x + off];
        __syncthreads();
    }
    if (threadIdx.x == 0) {
        float fin = red[0] + bq[0];
        float z   = __cosf(fin * qlw[0]) * __cosf(qp[0]);
        fgate[b]  = sigmoidf_(z);
    }
}

// ---------------------------------------------------------------------------
// Per-step fused kernel.
// Grid: 64 blocks (one 16-column n-tile each). Block: 96 threads = 3 waves,
// wave w = gate w (i/u/o). Each wave carries 8 M-tile accumulators (M=128)
// over K=2048; A k-chunks staged in LDS, weights streamed once.
// Epilogue: gates exchanged through LDS, cell update fused.
// ---------------------------------------------------------------------------
__global__ __launch_bounds__(96)
void qlstm_step(const bf16_t* __restrict__ combA,   // (128 x 2048) read
                bf16_t* __restrict__ combB,         // h-half written (next step)
                const bf16_t* __restrict__ Wb,      // (3 x 1024 x 2048)
                const float* __restrict__ bi, const float* __restrict__ bu,
                const float* __restrict__ bo,
                const float* __restrict__ fgate,    // (128)
                float* __restrict__ c,              // (128 x 1024) in/out
                float* __restrict__ h,              // (128 x 1024) out
                float* __restrict__ out_t) {        // d_out + t*B*DH
    __shared__ bf16_t ldsA[BATCH * A_STRIDE];       // 10240 B
    __shared__ float  ldsP[3 * P_GATE];             // 26112 B

    const int tid  = threadIdx.x;
    const int lane = tid & 31;
    const int gate = tid >> 5;         // 0..2 = i/u/o
    const int llo  = lane & 15;        // N within tile / row within M-tile
    const int lhi  = lane >> 4;        // half-wave selector

    const int n0 = blockIdx.x * 16;    // hidden-dim tile base

    // B fragment addressing: weight row n0+llo, 16 contiguous K per lane.
    const bf16_t* wr = Wb + (size_t)gate * DH * CIN +
                       (size_t)(n0 + llo) * CIN + 16 * lhi;

    v8f acc[8];
#pragma unroll
    for (int mt = 0; mt < 8; ++mt) acc[mt] = (v8f){};

    for (int kc = 0; kc < CIN; kc += 32) {
        // Cooperatively stage A chunk (128 rows x 32 bf16) into LDS.
#pragma unroll
        for (int s = 0; s < 6; ++s) {
            int seg = tid + s * 96;                // 0..575, want < 512
            if (seg < 512) {
                int row = seg >> 2, half = seg & 3;
                *(v8bf*)(ldsA + row * A_STRIDE + half * 8) =
                    *(const v8bf*)(combA + (size_t)row * CIN + kc + half * 8);
            }
        }
        __syncthreads();

        v16bf fb = *(const v16bf*)(wr + kc);       // weight fragment (global)

#pragma unroll
        for (int mt = 0; mt < 8; ++mt) {
            const bf16_t* lrow = ldsA + (mt * 16 + llo) * A_STRIDE + 16 * lhi;
            v8bf alo = *(const v8bf*)(lrow);
            v8bf ahi = *(const v8bf*)(lrow + 8);
            v16bf a  = __builtin_shufflevector(alo, ahi,
                           0, 1, 2, 3, 4, 5, 6, 7,
                           8, 9, 10, 11, 12, 13, 14, 15);
            acc[mt] = __builtin_amdgcn_wmma_f32_16x16x32_bf16(
                          false, a, false, fb, (short)0, acc[mt], false, false);
        }
        __syncthreads();
    }

    // C/D layout: element r of acc[mt] -> batch b = mt*16 + r + 8*lhi, N = llo.
    // Exchange pre-activations across the 3 gate waves via LDS.
#pragma unroll
    for (int mt = 0; mt < 8; ++mt) {
#pragma unroll
        for (int r = 0; r < 8; ++r) {
            int b = mt * 16 + r + 8 * lhi;
            ldsP[gate * P_GATE + b * P_STRIDE + llo] = acc[mt][r];
        }
    }
    __syncthreads();

    // Fused LSTM cell update for the 128 x 16 tile.
    for (int idx = tid; idx < BATCH * 16; idx += 96) {
        const int b  = idx >> 4;
        const int nl = idx & 15;
        const int n  = n0 + nl;
        const float pi = ldsP[0 * P_GATE + b * P_STRIDE + nl] + bi[n];
        const float pu = ldsP[1 * P_GATE + b * P_STRIDE + nl] + bu[n];
        const float po = ldsP[2 * P_GATE + b * P_STRIDE + nl] + bo[n];
        const float iv = sigmoidf_(pi);
        const float gv = tanhf(pu);
        const float ov = sigmoidf_(po);
        const size_t o = (size_t)b * DH + n;
        const float cn = fgate[b] * c[o] + iv * gv;
        const float hn = ov * tanhf(cn);
        c[o]     = cn;
        h[o]     = hn;
        out_t[o] = hn;
        combB[(size_t)b * CIN + DIN + n] = (bf16_t)hn;
    }
}

// ---------------------------------------------------------------------------
// Final: copy hx, cx into the tail of d_out.
// ---------------------------------------------------------------------------
__global__ void qlstm_tail(const float* __restrict__ h,
                           const float* __restrict__ c,
                           float* __restrict__ out) {
    int i = blockIdx.x * blockDim.x + threadIdx.x;
    if (i < BATCH * DH) {
        const size_t HX = (size_t)SEQ * BATCH * DH;       // 67108864
        out[HX + i]              = h[i];
        out[HX + BATCH * DH + i] = c[i];
    }
}

// ---------------------------------------------------------------------------
extern "C" void kernel_launch(void* const* d_in, const int* in_sizes, int n_in,
                              void* d_out, int out_size, void* d_ws,
                              size_t ws_size, hipStream_t stream) {
    (void)in_sizes; (void)n_in; (void)out_size; (void)ws_size;

    const float* x   = (const float*)d_in[0];
    const float* Wi  = (const float*)d_in[1];
    const float* bi  = (const float*)d_in[2];
    const float* Wu  = (const float*)d_in[3];
    const float* bu  = (const float*)d_in[4];
    const float* Wo  = (const float*)d_in[5];
    const float* bo  = (const float*)d_in[6];
    const float* Wq  = (const float*)d_in[7];
    const float* bq  = (const float*)d_in[8];
    const float* qlw = (const float*)d_in[9];
    const float* qp  = (const float*)d_in[10];
    float* out = (float*)d_out;

    // Workspace layout (~14.7 MB total).
    const size_t WB_BYTES   = (size_t)3 * DH * CIN * sizeof(bf16_t); // 12 MB
    const size_t COMB_BYTES = (size_t)BATCH * CIN * sizeof(bf16_t);  // 512 KB
    const size_t HC_BYTES   = (size_t)BATCH * DH * sizeof(float);    // 512 KB
    char* ws = (char*)d_ws;
    bf16_t* Wb    = (bf16_t*)(ws);
    bf16_t* comb0 = (bf16_t*)(ws + WB_BYTES);
    bf16_t* comb1 = (bf16_t*)(ws + WB_BYTES + COMB_BYTES);
    float*  c     = (float*) (ws + WB_BYTES + 2 * COMB_BYTES);
    float*  h     = (float*) (ws + WB_BYTES + 2 * COMB_BYTES + HC_BYTES);
    float*  fg    = (float*) (ws + WB_BYTES + 2 * COMB_BYTES + 2 * HC_BYTES);

    qlstm_cvt_w<<<4096, 256, 0, stream>>>(Wi, Wu, Wo, Wb);
    qlstm_init<<<512, 256, 0, stream>>>(c, h, comb0);

    for (int t = 0; t < SEQ; ++t) {
        bf16_t* cA = (t & 1) ? comb1 : comb0;   // read this step
        bf16_t* cB = (t & 1) ? comb0 : comb1;   // h written for next step
        const float* xt = x + (size_t)t * BATCH * DIN;
        qlstm_cvt_x<<<512, 256, 0, stream>>>(xt, cA);
        qlstm_qgate<<<BATCH, 256, 0, stream>>>(xt, h, Wq, bq, qlw, qp, fg);
        qlstm_step<<<DH / 16, 96, 0, stream>>>(
            cA, cB, Wb, bi, bu, bo, fg, c, h, out + (size_t)t * BATCH * DH);
    }

    qlstm_tail<<<512, 256, 0, stream>>>(h, c, out);
}
